// EquivariantGNNBlock_12128987644091
// MI455X (gfx1250) — matrix operand order repaired and checked by reading, not compiled, not run on hardware
//
#include <hip/hip_runtime.h>
#include <hip/hip_bf16.h>

// ---------------------------------------------------------------------------
// EquivariantGNNBlock for MI455X (gfx1250): f16 WMMA pipeline, fused paths.
// B=64, N=64, E=4032, H=256, A=4, FI=517 (padded to 544), hidden=256.
// ---------------------------------------------------------------------------

#define BB    64
#define NN    64
#define EE    4032
#define AA    4
#define HH    256
#define FIP   544              // 517 padded to multiple of 32
#define FSTR  552              // feat LDS row stride (halves), 16B-aligned rows
#define MSTR  264              // 256-wide intermediate tile stride (halves)
#define NSTR  520              // 512-wide node-concat tile stride (halves)
#define SCALE_ 15.0f

typedef _Float16 v16h __attribute__((ext_vector_type(16)));
typedef _Float16 v8h  __attribute__((ext_vector_type(8)));
typedef float    v8f  __attribute__((ext_vector_type(8)));

__device__ __forceinline__ float sigmoid_(float x) { return 1.0f / (1.0f + __expf(-x)); }
__device__ __forceinline__ float silu_(float x)    { return x * sigmoid_(x); }

// ---- WMMA fragment loaders -------------------------------------------------
// A (16x32 f16) from LDS, row-major with stride strA halves.
// Layout: lanes 0-15 row M=lane, halves {K0..7, K16..23};
//         lanes 16-31 row M=lane-16, halves {K8..15, K24..31}.
__device__ __forceinline__ v16h load_a_frag(const _Float16* Alds, int strA, int lane, int kb) {
    int M     = lane & 15;
    int khalf = lane >> 4;
    const _Float16* p = Alds + (size_t)M * strA + kb * 32 + khalf * 8;
    v8h lo = *(const v8h*)(p);
    v8h hi = *(const v8h*)(p + 16);
    return __builtin_shufflevector(lo, hi, 0,1,2,3,4,5,6,7,8,9,10,11,12,13,14,15);
}

// B (32x16 f16) from globally packed weights: each lane's 16 halves contiguous.
__device__ __forceinline__ v16h load_b_frag(const _Float16* Wp, int ntiles, int nt, int kb, int lane) {
    return *(const v16h*)(Wp + ((((size_t)kb * ntiles + nt) * 32) + lane) * 16);
}

#define WMMA_F16(A, Bv, C) \
    __builtin_amdgcn_wmma_f32_16x16x32_f16(false, (A), false, (Bv), (short)0, (C), false, false)

// Epilogue: bias + SiLU + f16 store of two 16x16 tiles.
__device__ __forceinline__ void store2tiles(const v8f& a0, const v8f& a1,
                                            const float* bias, _Float16* Dst, int strD,
                                            int nt0, int lane) {
    const int nl = lane & 15;
    const int hi = lane >> 4;
    const float b0 = bias[nt0 * 16 + nl];
    const float b1 = bias[(nt0 + 1) * 16 + nl];
#pragma unroll
    for (int r = 0; r < 8; ++r) {
        int M = r + hi * 8;
        Dst[(size_t)M * strD + nt0 * 16 + nl]       = (_Float16)silu_(a0[r] + b0);
        Dst[(size_t)M * strD + (nt0 + 1) * 16 + nl] = (_Float16)silu_(a1[r] + b1);
    }
}

// Fused layer-1: one A stream (feat), two weight streams (message+coord).
// Per k-step: 2 ds_load_b128 + 8 global_load_b128 -> 4 WMMAs.
__device__ __forceinline__ void gemm1_fused(const _Float16* A, int strA,
                                            const _Float16* Wm, const _Float16* Wc, int kbs,
                                            const float* bm, const float* bc,
                                            _Float16* Dm, _Float16* Dc, int strD,
                                            int wid, int lane) {
    const int nt0 = wid * 2;
    v8f am0 = {}, am1 = {}, ac0 = {}, ac1 = {};
    for (int kb = 0; kb < kbs; ++kb) {
        if (kb + 1 < kbs) {
            __builtin_prefetch(Wm + (((size_t)(kb + 1) * 16 + nt0) * 32) * 16, 0, 1);
            __builtin_prefetch(Wc + (((size_t)(kb + 1) * 16 + nt0) * 32) * 16, 0, 1);
        }
        v16h af  = load_a_frag(A, strA, lane, kb);
        v16h bm0 = load_b_frag(Wm, 16, nt0,     kb, lane);
        v16h bm1 = load_b_frag(Wm, 16, nt0 + 1, kb, lane);
        v16h bc0 = load_b_frag(Wc, 16, nt0,     kb, lane);
        v16h bc1 = load_b_frag(Wc, 16, nt0 + 1, kb, lane);
        am0 = WMMA_F16(af, bm0, am0);
        am1 = WMMA_F16(af, bm1, am1);
        ac0 = WMMA_F16(af, bc0, ac0);
        ac1 = WMMA_F16(af, bc1, ac1);
    }
    store2tiles(am0, am1, bm, Dm, strD, nt0, lane);
    store2tiles(ac0, ac1, bc, Dc, strD, nt0, lane);
}

// Fused layer-2: two independent A streams, two weight streams.
__device__ __forceinline__ void gemm2_fused(const _Float16* Am, const _Float16* Ac, int strA,
                                            const _Float16* Wm, const _Float16* Wc, int kbs,
                                            const float* bm, const float* bc,
                                            _Float16* Dm, _Float16* Dc, int strD,
                                            int wid, int lane) {
    const int nt0 = wid * 2;
    v8f am0 = {}, am1 = {}, ac0 = {}, ac1 = {};
    for (int kb = 0; kb < kbs; ++kb) {
        v16h afm = load_a_frag(Am, strA, lane, kb);
        v16h afc = load_a_frag(Ac, strA, lane, kb);
        v16h bm0 = load_b_frag(Wm, 16, nt0,     kb, lane);
        v16h bm1 = load_b_frag(Wm, 16, nt0 + 1, kb, lane);
        v16h bc0 = load_b_frag(Wc, 16, nt0,     kb, lane);
        v16h bc1 = load_b_frag(Wc, 16, nt0 + 1, kb, lane);
        am0 = WMMA_F16(afm, bm0, am0);
        am1 = WMMA_F16(afm, bm1, am1);
        ac0 = WMMA_F16(afc, bc0, ac0);
        ac1 = WMMA_F16(afc, bc1, ac1);
    }
    store2tiles(am0, am1, bm, Dm, strD, nt0, lane);
    store2tiles(ac0, ac1, bc, Dc, strD, nt0, lane);
}

// Single GEMM pair (node kernel layer 1).
__device__ __forceinline__ void gemm2tiles(const _Float16* A, int strA,
                                           const _Float16* Wp, int kbs,
                                           const float* bias,
                                           _Float16* Dst, int strD,
                                           int wid, int lane) {
    const int nt0 = wid * 2;
    v8f acc0 = {}, acc1 = {};
    for (int kb = 0; kb < kbs; ++kb) {
        if (kb + 1 < kbs)
            __builtin_prefetch(Wp + (((size_t)(kb + 1) * 16 + nt0) * 32) * 16, 0, 1);
        v16h af = load_a_frag(A, strA, lane, kb);
        v16h b0 = load_b_frag(Wp, 16, nt0,     kb, lane);
        v16h b1 = load_b_frag(Wp, 16, nt0 + 1, kb, lane);
        acc0 = WMMA_F16(af, b0, acc0);
        acc1 = WMMA_F16(af, b1, acc1);
    }
    store2tiles(acc0, acc1, bias, Dst, strD, nt0, lane);
}

// ---- small utility kernels -------------------------------------------------
__global__ void zero_f32(float* p, long n) {
    long i = (long)blockIdx.x * blockDim.x + threadIdx.x;
    if (i < n) p[i] = 0.0f;
}

__global__ void cvt_h_f16(const float* __restrict__ h, _Float16* __restrict__ h16, long n) {
    long i = (long)blockIdx.x * blockDim.x + threadIdx.x;
    if (i < n) h16[i] = (_Float16)h[i];
}

// Pack a f32 weight [K x 256] into WMMA-B-fragment order, K padded to Kp.
__global__ void pack_w(const float* __restrict__ src, _Float16* __restrict__ dst, int K, int Kp) {
    long total = (long)Kp * 256;
    long p = (long)blockIdx.x * blockDim.x + threadIdx.x;
    if (p >= total) return;
    int t  = (int)(p & 15);
    int l  = (int)((p >> 4) & 31);
    long r = p >> 9;                 // 512 halves per (kb, ntile) chunk
    int nt = (int)(r & 15);          // 16 n-tiles
    int kb = (int)(r >> 4);
    int k = kb * 32 + (l >> 4) * 8 + (t & 7) + (t >> 3) * 16;
    int n = nt * 16 + (l & 15);
    dst[p] = (k < K) ? (_Float16)src[(size_t)k * 256 + n] : (_Float16)0.0f;
}

// ---- edge kernel: one block = one 16-edge tile -----------------------------
__global__ __launch_bounds__(256) void edge_kernel(
    const float* __restrict__ x, const float* __restrict__ a,
    const int* __restrict__ eidx, const float* __restrict__ emask,
    const _Float16* __restrict__ h16,
    const _Float16* __restrict__ W1p, const _Float16* __restrict__ W2p,
    const _Float16* __restrict__ Wx1p, const _Float16* __restrict__ Wx2p,
    const float* __restrict__ be1, const float* __restrict__ be2,
    const float* __restrict__ Wa, const float* __restrict__ ba,
    const float* __restrict__ bx1, const float* __restrict__ bx2,
    const float* __restrict__ Wx3,
    float* __restrict__ em_agg, float* __restrict__ x_agg)
{
    __shared__ __align__(32) _Float16 featA[16 * FSTR];
    __shared__ __align__(32) _Float16 mAm[16 * MSTR];   // layer-1 out, message
    __shared__ __align__(32) _Float16 mAc[16 * MSTR];   // layer-1 out, coord
    __shared__ __align__(32) _Float16 mBm[16 * MSTR];   // layer-2 out, message (= m)
    __shared__ __align__(32) _Float16 mBc[16 * MSTR];   // layer-2 out, coord   (= u)
    __shared__ int   giT[16];
    __shared__ int   gjT[16];
    __shared__ float diffT[16][3];
    __shared__ float dT[16];
    __shared__ float d2T[16];
    __shared__ float eT[16];   // sigmoid gate
    __shared__ float wT[16];   // tanh coord weight
    __shared__ float red[64];

    const int tid  = threadIdx.x;
    const int lane = tid & 31;
    const int wid  = tid >> 5;
    const long e0  = (long)blockIdx.x * 16;    // EE % 16 == 0 -> tile never straddles batches

    if (tid < 16) {
        long ge = e0 + tid;
        int b = (int)(ge / EE);
        int ii = eidx[ge * 2 + 0];
        int jj = eidx[ge * 2 + 1];
        int gi = b * NN + ii;
        int gj = b * NN + jj;
        giT[tid] = gi;  gjT[tid] = gj;
        float em = emask[ge];
        float d2 = 0.0f;
#pragma unroll
        for (int k = 0; k < 3; ++k) {
            float df = (x[(size_t)gi * 3 + k] - x[(size_t)gj * 3 + k]) * em;
            diffT[tid][k] = df;
            d2 += df * df;
        }
        d2T[tid] = d2;
        dT[tid]  = sqrtf(d2);
    }
    __syncthreads();

    // Build padded f16 feature tile [16 x 544]: [h_i | h_j | d^2 | a | 0-pad]
    // Vector part: cols 0..511 in 16-byte chunks (global_load_b128 -> ds_store_b128).
    for (int idx = tid; idx < 16 * 64; idx += 256) {
        int m  = idx >> 6;
        int c  = (idx & 63) * 8;
        const _Float16* src = (c < 256)
            ? (h16 + (size_t)giT[m] * HH + c)
            : (h16 + (size_t)gjT[m] * HH + (c - 256));
        *(v8h*)(featA + (size_t)m * FSTR + c) = *(const v8h*)src;
    }
    // Scalar tail: cols 512..551 (d^2, edge attrs, zero pad).
    for (int idx = tid; idx < 16 * (FSTR - 512); idx += 256) {
        int m = idx / (FSTR - 512);
        int c = 512 + idx % (FSTR - 512);
        long ge = e0 + m;
        _Float16 v = (_Float16)0.0f;
        if (c == 512)     v = (_Float16)d2T[m];
        else if (c < 517) v = (_Float16)a[ge * AA + (c - 513)];
        featA[(size_t)m * FSTR + c] = v;
    }
    __syncthreads();

    // Layer 1 (fused message+coord): K=544, shared A fragments.
    gemm1_fused(featA, FSTR, W1p, Wx1p, FIP / 32, be1, bx1, mAm, mAc, MSTR, wid, lane);
    __syncthreads();
    // Layer 2 (fused): K=256.
    gemm2_fused(mAm, mAc, MSTR, W2p, Wx2p, 8, be2, bx2, mBm, mBc, MSTR, wid, lane);
    __syncthreads();

    // Gate dot products: wave0 -> m.Wa, wave1 -> u.Wx3 (both halves per edge).
    if (tid < 64) {
        int m  = tid & 15;
        int hf = (tid >> 4) & 1;
        const _Float16* src = (tid < 32) ? mBm : mBc;
        const float*    w   = (tid < 32) ? Wa  : Wx3;
        float s = 0.0f;
        for (int n = hf * 128; n < hf * 128 + 128; ++n)
            s += (float)src[(size_t)m * MSTR + n] * w[n];
        red[tid] = s;
    }
    __syncthreads();
    if (tid < 16)                 eT[tid]      = sigmoid_(red[tid] + red[tid + 16] + ba[0]);
    else if (tid < 32)            wT[tid - 16] = tanhf(red[tid + 16] + red[tid + 32]) * SCALE_;
    __syncthreads();

    // Scatter e*m into em_agg (segment-sum over source nodes).
    for (int idx = tid; idx < 16 * 256; idx += 256) {
        int m = idx >> 8, n = idx & 255;
        float v = eT[m] * (float)mBm[(size_t)m * MSTR + n];
        atomicAdd(&em_agg[(size_t)giT[m] * 256 + n], v);
    }
    // Scatter coordinate update.
    if (tid < 48) {
        int m = tid / 3, ax = tid % 3;
        float v = wT[m] * diffT[m][ax] / (dT[m] + 1.0f);
        atomicAdd(&x_agg[(size_t)giT[m] * 3 + ax], v);
    }
}

// ---- node kernel: one block = one 16-node tile -----------------------------
__global__ __launch_bounds__(256) void node_kernel(
    const float* __restrict__ h, const _Float16* __restrict__ h16,
    const float* __restrict__ em_agg,
    const _Float16* __restrict__ Wh1p, const _Float16* __restrict__ Wh2p,
    const float* __restrict__ bh1, const float* __restrict__ bh2,
    const float* __restrict__ nmask,
    float* __restrict__ h_out)
{
    __shared__ __align__(32) _Float16 Atile[16 * NSTR];
    __shared__ __align__(32) _Float16 mA[16 * MSTR];

    const int tid  = threadIdx.x;
    const int lane = tid & 31;
    const int wid  = tid >> 5;
    const int g0   = blockIdx.x * 16;

    // concat(h, em_agg) as f16 [16 x 512]; h part vectorized.
    for (int idx = tid; idx < 16 * 32; idx += 256) {
        int m = idx >> 5;
        int c = (idx & 31) * 8;
        *(v8h*)(Atile + (size_t)m * NSTR + c) = *(const v8h*)(h16 + (size_t)(g0 + m) * HH + c);
    }
    for (int idx = tid; idx < 16 * (NSTR - 256); idx += 256) {
        int m = idx / (NSTR - 256);
        int c = 256 + idx % (NSTR - 256);
        _Float16 v = (_Float16)0.0f;
        if (c < 512) v = (_Float16)em_agg[(size_t)(g0 + m) * 256 + (c - 256)];
        Atile[(size_t)m * NSTR + c] = v;
    }
    __syncthreads();

    gemm2tiles(Atile, NSTR, Wh1p, 16, bh1, mA, MSTR, wid, lane);
    __syncthreads();

    // Layer 2: WMMA + residual + bias + mask, fp32 out.
    const int nt0 = wid * 2;
    v8f acc0 = {}, acc1 = {};
    for (int kb = 0; kb < 8; ++kb) {
        v16h af = load_a_frag(mA, MSTR, lane, kb);
        v16h b0 = load_b_frag(Wh2p, 16, nt0,     kb, lane);
        v16h b1 = load_b_frag(Wh2p, 16, nt0 + 1, kb, lane);
        acc0 = WMMA_F16(af, b0, acc0);
        acc1 = WMMA_F16(af, b1, acc1);
    }
    const int nl = lane & 15;
    const int hi = lane >> 4;
#pragma unroll
    for (int r = 0; r < 8; ++r) {
        int M = r + hi * 8;
        int g = g0 + M;
        float msk = nmask[g];
        int n0 = nt0 * 16 + nl;
        int n1 = (nt0 + 1) * 16 + nl;
        h_out[(size_t)g * HH + n0] = (acc0[r] + bh2[n0] + h[(size_t)g * HH + n0]) * msk;
        h_out[(size_t)g * HH + n1] = (acc1[r] + bh2[n1] + h[(size_t)g * HH + n1]) * msk;
    }
}

__global__ void xout_kernel(const float* __restrict__ x, const float* __restrict__ x_agg,
                            const float* __restrict__ nmask, float* __restrict__ x_out) {
    int i = blockIdx.x * blockDim.x + threadIdx.x;
    if (i < BB * NN * 3) x_out[i] = (x[i] + x_agg[i]) * nmask[i / 3];
}

// ---------------------------------------------------------------------------
extern "C" void kernel_launch(void* const* d_in, const int* in_sizes, int n_in,
                              void* d_out, int out_size, void* d_ws, size_t ws_size,
                              hipStream_t stream) {
    const float* x     = (const float*)d_in[0];
    const float* h     = (const float*)d_in[1];
    const float* a     = (const float*)d_in[2];
    const int*   eidx  = (const int*)  d_in[3];
    const float* nmask = (const float*)d_in[4];
    const float* emask = (const float*)d_in[5];
    const float* We1 = (const float*)d_in[6],  *be1 = (const float*)d_in[7];
    const float* We2 = (const float*)d_in[8],  *be2 = (const float*)d_in[9];
    const float* Wa  = (const float*)d_in[10], *ba  = (const float*)d_in[11];
    const float* Wh1 = (const float*)d_in[12], *bh1 = (const float*)d_in[13];
    const float* Wh2 = (const float*)d_in[14], *bh2 = (const float*)d_in[15];
    const float* Wx1 = (const float*)d_in[16], *bx1 = (const float*)d_in[17];
    const float* Wx2 = (const float*)d_in[18], *bx2 = (const float*)d_in[19];
    const float* Wx3 = (const float*)d_in[20];

    // workspace layout (bytes)
    char* ws = (char*)d_ws;
    _Float16* h16  = (_Float16*)(ws + 0);                    // 2,097,152
    _Float16* W1p  = (_Float16*)(ws + 2097152);              //   278,528 (544x256)
    _Float16* W2p  = (_Float16*)(ws + 2375680);              //   131,072
    _Float16* Wx1p = (_Float16*)(ws + 2506752);              //   278,528
    _Float16* Wx2p = (_Float16*)(ws + 2785280);              //   131,072
    _Float16* Wh1p = (_Float16*)(ws + 2916352);              //   262,144 (512x256)
    _Float16* Wh2p = (_Float16*)(ws + 3178496);              //   131,072
    float*    em_agg = (float*)(ws + 3309568);               // 4,194,304
    float*    x_agg  = (float*)(ws + 7503872);               //    49,152

    float* x_out = (float*)d_out;                 // [B*N*3]
    float* h_out = (float*)d_out + BB * NN * 3;   // [B*N*256]

    const long nEm = (long)BB * NN * 256 + (long)BB * NN * 3;  // contiguous zero range
    zero_f32<<<(int)((nEm + 255) / 256), 256, 0, stream>>>(em_agg, nEm);

    const long nH = (long)BB * NN * HH;
    cvt_h_f16<<<(int)((nH + 255) / 256), 256, 0, stream>>>(h, h16, nH);

    pack_w<<<(544 * 256) / 256, 256, 0, stream>>>(We1, W1p, 517, 544);
    pack_w<<<(256 * 256) / 256, 256, 0, stream>>>(We2, W2p, 256, 256);
    pack_w<<<(544 * 256) / 256, 256, 0, stream>>>(Wx1, Wx1p, 517, 544);
    pack_w<<<(256 * 256) / 256, 256, 0, stream>>>(Wx2, Wx2p, 256, 256);
    pack_w<<<(512 * 256) / 256, 256, 0, stream>>>(Wh1, Wh1p, 512, 512);
    pack_w<<<(256 * 256) / 256, 256, 0, stream>>>(Wh2, Wh2p, 256, 256);

    const int nEdgeTiles = (BB * EE) / 16;   // 16128
    edge_kernel<<<nEdgeTiles, 256, 0, stream>>>(x, a, eidx, emask, h16,
                                                W1p, W2p, Wx1p, Wx2p,
                                                be1, be2, Wa, ba, bx1, bx2, Wx3,
                                                em_agg, x_agg);

    const int nNodeTiles = (BB * NN) / 16;   // 256
    node_kernel<<<nNodeTiles, 256, 0, stream>>>(h, h16, em_agg, Wh1p, Wh2p,
                                                bh1, bh2, nmask, h_out);

    xout_kernel<<<(BB * NN * 3 + 255) / 256, 256, 0, stream>>>(x, x_agg, nmask, x_out);
}
